// RESA_19559281066402
// MI455X (gfx1250) — compile-verified, matrix operand order checked by reading, not run on hardware
//
#include <hip/hip_runtime.h>
#include <hip/hip_bf16.h>
#include <stddef.h>

// ---------------------------------------------------------------------------
// RESA directional aggregation for MI455X (gfx1250, wave32, WMMA).
//
// Each of 20 steps is, per (b,h) output row, a GEMM:
//     out[co,w] = sum_{ci,tap} W[co, tap*128+ci] * X[ci, colIdx(w,tap)]
// with M=128, N=96, K=1152, done with v_wmma_f32_16x16x32_bf16.
// K is ordered (tap major, ci minor) so every 32-wide K chunk sits inside a
// single tap => the B fragment is a plain strided read of the ci-contiguous
// staged row in LDS (no im2col).
//
// d_ws layout (assumed ws_size >= ~132 MB):
//   [0, 125829120)                    : ping buffer for y (fp32, 64*128*40*96)
//   [125829120, +5898240)             : bf16 weights pre-swizzled into WMMA
//                                       A-fragment order, 20 convs
// ---------------------------------------------------------------------------

typedef __attribute__((ext_vector_type(16))) __bf16 v16bf;
typedef __attribute__((ext_vector_type(8)))  __bf16 v8bf;
typedef __attribute__((ext_vector_type(8)))  float  v8f;

#define RESA_B   64
#define RESA_C   128
#define RESA_H   40
#define RESA_W   96
#define CI_STR   72                      // LDS ci-stride in halves (144 B, 16B aligned)
#define LDS_HALV (9 * RESA_W * CI_STR)   // 62,208 halves = 124,416 B (vertical worst case)
#define FRAG_HALVES_PER_CONV (8 * 36 * 512)   // 294,912 bf16 per conv
#define PING_BYTES ((size_t)RESA_B * RESA_C * RESA_H * RESA_W * 4)  // 125,829,120

// --------------------------------------------------------------------------
// Prologue: swizzle fp32 conv weights into the exact 16-bit A-matrix WMMA
// fragment layout (ISA 7.12.2): lane<16 holds M=lane, halves j: K = j (j<8),
// 8+j (j>=8); lanes>=16 hold same M with K offset +8.  K = kIter*32 + kk,
// remapped so K = tap*128 + ci.
// --------------------------------------------------------------------------
__global__ __launch_bounds__(256) void resa_swizzle_weights(
    const float* __restrict__ wd, const float* __restrict__ wu,
    const float* __restrict__ wr, const float* __restrict__ wl,
    __bf16* __restrict__ out)
{
    int gid = blockIdx.x * 256 + threadIdx.x;       // 20*8*36*512 = 2,949,120
    int j    = gid & 15;
    int lane = (gid >> 4) & 31;
    int rest = gid >> 9;                            // ((c*8)+t)*36 + kI
    int kI   = rest % 36;
    int r2   = rest / 36;
    int t    = r2 & 7;
    int c    = r2 >> 3;
    if (c >= 20) return;

    int kk  = (j & 7) + ((j >> 3) << 4) + ((lane >> 4) << 3);
    int K   = kI * 32 + kk;
    int tap = K >> 7;                               // K = tap*128 + ci
    int ci  = K & 127;
    int m   = t * 16 + (lane & 15);

    int dir = c / 5, it = c % 5;
    const float* wsrc = (dir == 0) ? wd : (dir == 1) ? wu : (dir == 2) ? wr : wl;
    // wd/wu: [5,C,C,1,9], wr/wl: [5,C,C,9,1] -- both flatten to [5][128][128][9]
    float v = wsrc[(((size_t)it * RESA_C + m) * RESA_C + ci) * 9 + tap];
    out[gid] = (__bf16)v;
}

// --------------------------------------------------------------------------
// One RESA step: dst = src + 2*relu(conv(roll(src))).
// grid = B*H blocks (one output row each), 256 threads = 8 waves.
// Wave w owns M-tile w (16 output channels) x 6 N-tiles (96 cols).
// Channels staged in two 64-ci halves to keep LDS at 124 KB.
// --------------------------------------------------------------------------
__global__ __launch_bounds__(256) void resa_step(
    const float* __restrict__ src, float* __restrict__ dst,
    const __bf16* __restrict__ wfrag,   // this conv's swizzled weights
    int vertical,                       // 0: conv along W (rows rolled along H)
                                        // 1: conv along H (cols rolled along W)
    int shift)                          // positive roll amount (mod H or W)
{
    __shared__ __bf16 sX[LDS_HALV];

    const int tid    = threadIdx.x;
    const int lane   = tid & 31;
    const int wave   = tid >> 5;          // == M-tile index, 0..7
    const int laneN  = lane & 15;
    const int hiHalf = lane >> 4;         // lane group (0: lanes 0-15, 1: 16-31)
    const int bh = blockIdx.x;
    const int b  = bh / RESA_H;
    const int h  = bh - b * RESA_H;

    const float* srcb = src + (size_t)b * RESA_C * RESA_H * RESA_W;

    v8f zero8 = {0.f, 0.f, 0.f, 0.f, 0.f, 0.f, 0.f, 0.f};
    v8f acc[6];
#pragma unroll
    for (int n = 0; n < 6; ++n) acc[n] = zero8;

    const int kk0 = hiHalf * 8;           // first kk of this lane group's runs

    for (int g = 0; g < 2; ++g) {         // channel halves: ci in [g*64, g*64+64)
        __syncthreads();                  // previous half fully consumed

        // ------------------------- stage into LDS -------------------------
        if (!vertical) {
            // Horizontal conv: one rolled row, padded cols [0..3],[100..103].
            for (int i = tid; i < 8 * 64; i += 256) {
                int c4 = i >> 6, cil = i & 63;
                int col = (c4 < 4) ? c4 : (96 + c4);
                sX[col * CI_STR + cil] = (__bf16)0.f;
            }
            int hsrc = h + shift; if (hsrc >= RESA_H) hsrc -= RESA_H;
            const float* row = srcb + ((size_t)(g * 64) * RESA_H + hsrc) * RESA_W;
            for (int i = tid; i < 64 * 96; i += 256) {
                int cil = i / 96, w = i - cil * 96;      // w fastest: coalesced
                float v = row[(size_t)cil * RESA_H * RESA_W + w];
                sX[(w + 4) * CI_STR + cil] = (__bf16)v;  // bank step 36: clean
            }
        } else {
            // Vertical conv: 9 h-rows (zero outside [0,H)), cols rolled by shift.
            for (int i = tid; i < 9 * 64 * 96; i += 256) {
                int tap = i / (64 * 96);
                int r   = i - tap * 64 * 96;
                int cil = r / 96, w = r - cil * 96;
                int hsrc = h + tap - 4;
                int wsrc = w + shift; if (wsrc >= RESA_W) wsrc -= RESA_W;
                float v = 0.f;
                if (hsrc >= 0 && hsrc < RESA_H)
                    v = srcb[(((size_t)(g * 64 + cil)) * RESA_H + hsrc) * RESA_W + wsrc];
                sX[(tap * 96 + w) * CI_STR + cil] = (__bf16)v;
            }
        }
        __syncthreads();

        // --------------------- GEMM over this K half ----------------------
        // k_iters for this half: kI = tap*4 + 2g + q, q in {0,1}
        for (int tap = 0; tap < 9; ++tap) {
#pragma unroll
            for (int q = 0; q < 2; ++q) {
                const int kI = tap * 4 + g * 2 + q;
                // A fragment: pre-swizzled, two coalesced 16B loads per lane.
                const __bf16* ap = wfrag + ((size_t)(wave * 36 + kI)) * 512 + lane * 16;
                v8bf a_lo = *reinterpret_cast<const v8bf*>(ap);
                v8bf a_hi = *reinterpret_cast<const v8bf*>(ap + 8);
                v16bf afrag = __builtin_shufflevector(
                    a_lo, a_hi, 0,1,2,3,4,5,6,7,8,9,10,11,12,13,14,15);

                const int cibL = q * 32;   // local ci base within the 64-ci half
#pragma unroll
                for (int n = 0; n < 6; ++n) {
                    int col    = n * 16 + laneN;
                    int colIdx = vertical ? (tap * 96 + col) : (col + tap);
                    // B 32x16 fragment: lane group selects kk runs {kk0..kk0+7}
                    // and {kk0+16..kk0+23}; ci contiguous => two ds_load_b128.
                    const __bf16* bp = &sX[colIdx * CI_STR + cibL + kk0];
                    v8bf b_lo = *reinterpret_cast<const v8bf*>(bp);
                    v8bf b_hi = *reinterpret_cast<const v8bf*>(bp + 16);
                    v16bf bfrag = __builtin_shufflevector(
                        b_lo, b_hi, 0,1,2,3,4,5,6,7,8,9,10,11,12,13,14,15);
                    acc[n] = __builtin_amdgcn_wmma_f32_16x16x32_bf16(
                        false, afrag, false, bfrag, (short)0, acc[n], false, false);
                }
            }
        }
    }

    // --------- epilogue: dst = src + 2*relu(acc), C/D layout 7.12.2 --------
    // VGPR r, lanes 0-15 -> M=r; lanes 16-31 -> M=r+8; N = lane&15.
#pragma unroll
    for (int n = 0; n < 6; ++n) {
        int col = n * 16 + laneN;
#pragma unroll
        for (int r = 0; r < 8; ++r) {
            int m = wave * 16 + r + hiHalf * 8;
            size_t idx = (((size_t)b * RESA_C + m) * RESA_H + h) * RESA_W + col;
            float v = acc[n][r];
            v = v > 0.f ? v : 0.f;
            dst[idx] = src[idx] + 2.0f * v;   // 64B-coalesced half-wave segments
        }
    }
}

// --------------------------------------------------------------------------
extern "C" void kernel_launch(void* const* d_in, const int* in_sizes, int n_in,
                              void* d_out, int out_size, void* d_ws, size_t ws_size,
                              hipStream_t stream)
{
    (void)in_sizes; (void)n_in; (void)out_size; (void)ws_size;

    const float* x  = (const float*)d_in[0];
    const float* wd = (const float*)d_in[1];
    const float* wu = (const float*)d_in[2];
    const float* wr = (const float*)d_in[3];
    const float* wl = (const float*)d_in[4];

    float*  out   = (float*)d_out;
    float*  ping  = (float*)d_ws;                                  // buffer 0
    __bf16* wfrag = (__bf16*)((char*)d_ws + PING_BYTES);           // bf16 weights

    // 1) Swizzle all 20 convs' weights into WMMA fragment order (bf16).
    resa_swizzle_weights<<<(20 * 8 * 36 * 512) / 256, 256, 0, stream>>>(
        wd, wu, wr, wl, wfrag);

    // 2) 20 dependent steps; ping-pong ws <-> out so step 20 ends in d_out.
    for (int step = 1; step <= 20; ++step) {
        int c   = step - 1;
        int dir = c / 5, it = c % 5;
        int vertical = (dir >= 2);
        int off = vertical ? (RESA_W >> (5 - it)) : (RESA_H >> (5 - it));
        int shift;
        if      (dir == 0) shift = off;              // roll(y,-off,H): hsrc=h+off
        else if (dir == 1) shift = RESA_H - off;     // roll(y,+off,H)
        else if (dir == 2) shift = off;              // roll(y,-off,W): wsrc=w+off
        else               shift = RESA_W - off;     // roll(y,+off,W)

        const float* s = (step == 1) ? x : ((step & 1) ? out : ping);
        float*       d = (step & 1) ? ping : out;

        resa_step<<<RESA_B * RESA_H, 256, 0, stream>>>(
            s, d, wfrag + (size_t)c * FRAG_HALVES_PER_CONV, vertical, shift);
    }
}
